// ModularMamba_14559939133722
// MI455X (gfx1250) — compile-verified, matrix operand order checked by reading, not run on hardware
//
#include <hip/hip_runtime.h>
#include <hip/hip_bf16.h>

#define BATCH   4
#define TLEN    1024
#define DMODEL  1024
#define DSTATE  16
#define VOCAB   128
#define TCHUNK  128
#define KC      128         // K-chunk staged in LDS per TDM transfer (64 rows x 128 = 16 KB)

typedef float  v8f   __attribute__((ext_vector_type(8)));
typedef __bf16 v16bf __attribute__((ext_vector_type(16)));
typedef __bf16 v8bf  __attribute__((ext_vector_type(8)));
typedef unsigned u32x4 __attribute__((ext_vector_type(4)));
typedef int      i32x4v __attribute__((ext_vector_type(4)));
typedef int      i32x8v __attribute__((ext_vector_type(8)));

__device__ __forceinline__ __bf16 to_bf16(float f) {
  union { float f; unsigned u; } v; v.f = f;
  unsigned r = v.u + 0x7FFFu + ((v.u >> 16) & 1u);   // round-to-nearest-even
  unsigned short h = (unsigned short)(r >> 16);
  __bf16 o; __builtin_memcpy(&o, &h, sizeof(o));
  return o;
}

// ---------------------------------------------------------------- embedding
__global__ void k_embed(const int* __restrict__ tok, const float* __restrict__ emb,
                        float* __restrict__ x, __bf16* __restrict__ xh) {
  int idx = blockIdx.x * blockDim.x + threadIdx.x;   // 0 .. B*T*D-1
  int row = idx >> 10;
  int d   = idx & (DMODEL - 1);
  float v = emb[tok[row] * DMODEL + d];
  x[idx]  = v;
  xh[idx] = to_bf16(v);
}

// ---------------------------------------------------------------- f32 -> bf16
__global__ void k_f32_to_bf16(const float* __restrict__ src, __bf16* __restrict__ dst, int n) {
  int i = blockIdx.x * blockDim.x + threadIdx.x;
  if (i < n) dst[i] = to_bf16(src[i]);
}

// ---------------------------------------------------------------- Bx / Cx projections
__global__ void k_bxcx(const float* __restrict__ x, const float* __restrict__ Wb,
                       const float* __restrict__ Wc, float* __restrict__ Bx,
                       float* __restrict__ Cx) {
  int wid  = threadIdx.x >> 5;
  int lane = threadIdx.x & 31;
  int row  = blockIdx.x * (blockDim.x >> 5) + wid;          // 0 .. 4095
  int s    = lane & 15;
  const float4* wrow = (const float4*)(((lane < 16) ? Wb : Wc) + s * DMODEL);
  const float4* xrow = (const float4*)(x + row * DMODEL);
  float acc = 0.f;
  for (int k = 0; k < DMODEL / 4; ++k) {
    float4 xv = xrow[k], wv = wrow[k];
    acc += xv.x * wv.x + xv.y * wv.y + xv.z * wv.z + xv.w * wv.w;
  }
  if (lane < 16) Bx[row * DSTATE + s] = acc;
  else           Cx[row * DSTATE + s] = acc;
}

// ---------------------------------------------------------------- fragment load
__device__ __forceinline__ v16bf load_frag(const __bf16* p) {
  v8bf lo = *(const v8bf*)p;
  v8bf hi = *(const v8bf*)(p + 16);
  v16bf r;
#pragma unroll
  for (int i = 0; i < 8; ++i) { r[i] = lo[i]; r[i + 8] = hi[i]; }
  return r;
}

// ---------------------------------------------------------------- TDM issue: 64 x KC bf16 tile -> LDS
// D# group0: count=1 | lds_addr | global_addr(57b) | type=2
// D# group1: data_size=2B, tensor 1024x1024, tile KCx64, dim0_stride=1024
__device__ __forceinline__ void tdm_issue_tile(const __bf16* Wh, int n0, int k0,
                                               unsigned lds_addr) {
  unsigned long long ga =
      (unsigned long long)(uintptr_t)(Wh + (size_t)n0 * DMODEL + k0);
  u32x4 g0;
  g0[0] = 1u;                                   // count=1, user descriptor
  g0[1] = lds_addr;                             // LDS byte address
  g0[2] = (unsigned)ga;                         // global_addr[31:0]
  g0[3] = (unsigned)(ga >> 32) | (2u << 30);    // global_addr[56:32] | type=2
  i32x8v g1;
  g1[0] = (int)(1u << 16);                                  // data_size = 2 bytes
  g1[1] = (int)((DMODEL & 0xFFFFu) << 16);                  // tensor_dim0 lo16
  g1[2] = (int)((DMODEL >> 16) | ((DMODEL & 0xFFFFu) << 16)); // dim0 hi | dim1 lo
  g1[3] = (int)((DMODEL >> 16) | ((unsigned)KC << 16));     // dim1 hi | tile_dim0
  g1[4] = 64;                                               // tile_dim1=64, tile_dim2=0
  g1[5] = DMODEL;                                           // dim0_stride lo32
  g1[6] = 0;
  g1[7] = 0;
  i32x4v z4 = {0, 0, 0, 0};
#if defined(__clang_major__) && (__clang_major__ >= 23)
  i32x8v z8 = {0, 0, 0, 0, 0, 0, 0, 0};
  __builtin_amdgcn_tensor_load_to_lds(g0, g1, z4, z4, z8, 0);
#else
  __builtin_amdgcn_tensor_load_to_lds(g0, g1, z4, z4, 0);
#endif
}

// ---------------------------------------------------------------- gate GEMM (WMMA bf16, TDM-fed)
// Block = 8 waves -> 128x64 macro-tile. Wg tile (64 rows x KC) staged in LDS by TDM,
// double-buffered; wave 0 issues chunk c+1 while everyone computes chunk c.
__global__ void __launch_bounds__(256)
k_gate_gemm(const __bf16* __restrict__ Xh, const __bf16* __restrict__ Wh,
            const float* __restrict__ bg, float* __restrict__ G) {
  const int K = DMODEL, N = DMODEL;
  __shared__ __bf16 sW[2][64 * KC];           // 2 x 16 KB
  int tid  = threadIdx.x;
  int wid  = tid >> 5;
  int lane = tid & 31;
  int mb = blockIdx.x >> 4;                   // 32 m-blocks of 128 rows
  int ng = blockIdx.x & 15;                   // 16 n-groups of 64 cols
  int m0 = mb * 128 + wid * 16;
  int n0 = ng * 64;
  int r  = lane & 15;
  int kb = (lane >> 4) * 8;
  const __bf16* pa = Xh + (size_t)(m0 + r) * K + kb;
  unsigned lds0 = (unsigned)(uintptr_t)&sW[0][0];

  if (wid == 0) {
    tdm_issue_tile(Wh, n0, 0, lds0);
    __builtin_amdgcn_s_wait_tensorcnt(0);
  }
  v8f a0 = {0,0,0,0,0,0,0,0}, a1 = a0, a2 = a0, a3 = a0;
  __syncthreads();

  const int NCHUNK = K / KC;                  // 8
#pragma unroll
  for (int c = 0; c < NCHUNK; ++c) {
    if (wid == 0 && c + 1 < NCHUNK)
      tdm_issue_tile(Wh, n0, (c + 1) * KC,
                     lds0 + (unsigned)(((c + 1) & 1) * (64 * KC * 2)));
    const __bf16* sb = &sW[c & 1][0];
#pragma unroll
    for (int ks = 0; ks < KC; ks += 32) {
      if (c * KC + ks + 64 <= K) __builtin_prefetch(pa + c * KC + ks + 64, 0, 1);
      v16bf a  = load_frag(pa + c * KC + ks);
      v16bf b0 = load_frag(sb + ( 0 + r) * KC + ks + kb);
      v16bf b1 = load_frag(sb + (16 + r) * KC + ks + kb);
      v16bf b2 = load_frag(sb + (32 + r) * KC + ks + kb);
      v16bf b3 = load_frag(sb + (48 + r) * KC + ks + kb);
      a0 = __builtin_amdgcn_wmma_f32_16x16x32_bf16(false, a, false, b0, (short)0, a0, false, false);
      a1 = __builtin_amdgcn_wmma_f32_16x16x32_bf16(false, a, false, b1, (short)0, a1, false, false);
      a2 = __builtin_amdgcn_wmma_f32_16x16x32_bf16(false, a, false, b2, (short)0, a2, false, false);
      a3 = __builtin_amdgcn_wmma_f32_16x16x32_bf16(false, a, false, b3, (short)0, a3, false, false);
    }
    if (wid == 0 && c + 1 < NCHUNK)
      __builtin_amdgcn_s_wait_tensorcnt(0);
    __syncthreads();
  }

  // C/D layout: VGPR v, lane L -> M = v + (L>>4)*8, Nloc = L&15
  int mbase = m0 + (lane >> 4) * 8;
#pragma unroll
  for (int t = 0; t < 4; ++t) {
    v8f acc = (t == 0) ? a0 : (t == 1) ? a1 : (t == 2) ? a2 : a3;
    int n = n0 + t * 16 + r;
    float bias = bg[n];
#pragma unroll
    for (int v = 0; v < 8; ++v) {
      float val = acc[v] + bias;
      G[(size_t)(mbase + v) * N + n] = 1.f / (1.f + __expf(-val));
    }
  }
}

// ---------------------------------------------------------------- sequential SSM scan
__global__ void __launch_bounds__(256)
k_scan(const float* __restrict__ x, const float* __restrict__ Bx,
       const float* __restrict__ Cx, const float* __restrict__ A_log,
       const float* __restrict__ Dv, float* __restrict__ y) {
  __shared__ float sB[TCHUNK * DSTATE];
  __shared__ float sC[TCHUNK * DSTATE];
  int b = blockIdx.x >> 2;
  int d = ((blockIdx.x & 3) << 8) + threadIdx.x;
  float Ae[DSTATE], h[DSTATE];
#pragma unroll
  for (int s = 0; s < DSTATE; ++s) { Ae[s] = __expf(A_log[d * DSTATE + s]); h[s] = 0.f; }
  float Dd = Dv[d];
  const float* xp = x + (size_t)b * TLEN * DMODEL + d;
  float*       yp = y + (size_t)b * TLEN * DMODEL + d;
  const float* bp = Bx + (size_t)b * TLEN * DSTATE;
  const float* cp = Cx + (size_t)b * TLEN * DSTATE;
  for (int tc = 0; tc < TLEN; tc += TCHUNK) {
    __syncthreads();
    for (int j = threadIdx.x; j < TCHUNK * DSTATE; j += blockDim.x) {
      sB[j] = bp[tc * DSTATE + j];
      sC[j] = cp[tc * DSTATE + j];
    }
    __syncthreads();
    for (int tt = 0; tt < TCHUNK; ++tt) {
      int t = tc + tt;
      float xv = xp[t * DMODEL];
      float yv = 0.f;
#pragma unroll
      for (int s = 0; s < DSTATE; ++s) {
        h[s] = h[s] * Ae[s] + sB[tt * DSTATE + s] * xv;
        yv  += h[s] * sC[tt * DSTATE + s];
      }
      yp[t * DMODEL] = yv + Dd * xv;
    }
  }
}

// ---------------------------------------------------------------- gate mix + LayerNorm (in-place)
__global__ void __launch_bounds__(256)
k_gate_ln(float* __restrict__ x, const float* __restrict__ y,
          const float* __restrict__ g, const float* __restrict__ lnw,
          const float* __restrict__ lnb, __bf16* __restrict__ xh) {
  __shared__ float ssum[256];
  __shared__ float ssq[256];
  int row = blockIdx.x;
  int tid = threadIdx.x;
  int base = row * DMODEL + tid * 4;
  float4 xv = *(const float4*)(x + base);
  float4 yv = *(const float4*)(y + base);
  float4 gv = *(const float4*)(g + base);
  float o0 = gv.x * yv.x + (1.f - gv.x) * xv.x;
  float o1 = gv.y * yv.y + (1.f - gv.y) * xv.y;
  float o2 = gv.z * yv.z + (1.f - gv.z) * xv.z;
  float o3 = gv.w * yv.w + (1.f - gv.w) * xv.w;
  ssum[tid] = o0 + o1 + o2 + o3;
  ssq[tid]  = o0 * o0 + o1 * o1 + o2 * o2 + o3 * o3;
  __syncthreads();
  for (int off = 128; off > 0; off >>= 1) {
    if (tid < off) { ssum[tid] += ssum[tid + off]; ssq[tid] += ssq[tid + off]; }
    __syncthreads();
  }
  float mu  = ssum[0] * (1.f / DMODEL);
  float var = ssq[0] * (1.f / DMODEL) - mu * mu;
  float inv = rsqrtf(var + 1e-5f);
  float4 wv = *(const float4*)(lnw + tid * 4);
  float4 bv = *(const float4*)(lnb + tid * 4);
  float r0 = (o0 - mu) * inv * wv.x + bv.x;
  float r1 = (o1 - mu) * inv * wv.y + bv.y;
  float r2 = (o2 - mu) * inv * wv.z + bv.z;
  float r3 = (o3 - mu) * inv * wv.w + bv.w;
  float4 out = {r0, r1, r2, r3};
  *(float4*)(x + base) = out;
  union { __bf16 h[4]; unsigned long long u; } p;
  p.h[0] = to_bf16(r0); p.h[1] = to_bf16(r1); p.h[2] = to_bf16(r2); p.h[3] = to_bf16(r3);
  *(unsigned long long*)(xh + base) = p.u;
}

// ---------------------------------------------------------------- mean over T
__global__ void k_mean(const float* __restrict__ x, float* __restrict__ agg) {
  int b = blockIdx.x >> 2;
  int d = ((blockIdx.x & 3) << 8) + threadIdx.x;
  const float* p = x + (size_t)b * TLEN * DMODEL + d;
  float acc = 0.f;
  for (int t = 0; t < TLEN; ++t) acc += p[t * DMODEL];
  agg[b * DMODEL + d] = acc * (1.f / TLEN);
}

// ---------------------------------------------------------------- classifier
__global__ void k_logits(const float* __restrict__ agg, const float* __restrict__ W,
                         const float* __restrict__ bias, float* __restrict__ out) {
  int idx = blockIdx.x * blockDim.x + threadIdx.x;   // 0 .. 511
  int b = idx >> 7, v = idx & (VOCAB - 1);
  const float4* ap = (const float4*)(agg + b * DMODEL);
  const float4* wp = (const float4*)(W + v * DMODEL);
  float acc = 0.f;
  for (int k = 0; k < DMODEL / 4; ++k) {
    float4 a = ap[k], w = wp[k];
    acc += a.x * w.x + a.y * w.y + a.z * w.z + a.w * w.w;
  }
  out[idx] = acc + bias[v];
}

// ================================================================ launch
extern "C" void kernel_launch(void* const* d_in, const int* in_sizes, int n_in,
                              void* d_out, int out_size, void* d_ws, size_t ws_size,
                              hipStream_t stream) {
  const int*   tok   = (const int*)d_in[0];
  const float* emb   = (const float*)d_in[1];
  const float* cls_w = (const float*)d_in[2];
  const float* cls_b = (const float*)d_in[3];

  char* ws = (char*)d_ws;
  float*  X0  = (float*)(ws);                                   // 16 MB (in-place across layers)
  float*  Y   = (float*)(ws + (16ull << 20));                   // 16 MB
  float*  G   = (float*)(ws + (32ull << 20));                   // 16 MB
  __bf16* XH  = (__bf16*)(ws + (48ull << 20));                  //  8 MB
  __bf16* WGH = (__bf16*)(ws + (56ull << 20));                  //  2 MB
  float*  BX  = (float*)(ws + (58ull << 20));                   // 256 KB
  float*  CX  = (float*)(ws + (58ull << 20) + BATCH * TLEN * DSTATE * sizeof(float));
  float*  AGG = (float*)(ws + (59ull << 20));                   // 16 KB

  k_embed<<<(BATCH * TLEN * DMODEL) / 256, 256, 0, stream>>>(tok, emb, X0, XH);

  for (int L = 0; L < 2; ++L) {
    const float* A_log = (const float*)d_in[4 + L * 8 + 0];
    const float* Wb    = (const float*)d_in[4 + L * 8 + 1];
    const float* Wc    = (const float*)d_in[4 + L * 8 + 2];
    const float* Dv    = (const float*)d_in[4 + L * 8 + 3];
    const float* Wg    = (const float*)d_in[4 + L * 8 + 4];
    const float* bg    = (const float*)d_in[4 + L * 8 + 5];
    const float* lnw   = (const float*)d_in[4 + L * 8 + 6];
    const float* lnb   = (const float*)d_in[4 + L * 8 + 7];

    k_f32_to_bf16<<<(DMODEL * DMODEL) / 256, 256, 0, stream>>>(Wg, WGH, DMODEL * DMODEL);
    k_bxcx<<<(BATCH * TLEN) / 8, 256, 0, stream>>>(X0, Wb, Wc, BX, CX);
    k_gate_gemm<<<512, 256, 0, stream>>>(XH, WGH, bg, G);
    k_scan<<<16, 256, 0, stream>>>(X0, BX, CX, A_log, Dv, Y);
    k_gate_ln<<<BATCH * TLEN, 256, 0, stream>>>(X0, Y, G, lnw, lnb, XH);
  }

  k_mean<<<16, 256, 0, stream>>>(X0, AGG);
  k_logits<<<2, 256, 0, stream>>>(AGG, cls_w, cls_b, (float*)d_out);
}
